// BiMambaWrapper_2233382994678
// MI455X (gfx1250) — matrix operand order repaired and checked by reading, not compiled, not run on hardware
//
#include <hip/hip_runtime.h>
#include <math.h>

#define D_MODEL   1024
#define D_INNER   2048
#define HEADDIM   64
#define NHEADS    32
#define D_STATE   128
#define D_CONV    4
#define CONV_DIM  (D_INNER + 2*D_STATE)            // 2304
#define D_IN_PROJ (2*D_INNER + 2*D_STATE + NHEADS) // 4384
#define BSZ  4
#define SEQ  1024
#define ROWS (BSZ*SEQ)                             // 4096
#define EPSV 1e-5f

typedef __attribute__((ext_vector_type(16))) __bf16        bf16x16;
typedef __attribute__((ext_vector_type(8)))  float         f32x8;
typedef __attribute__((ext_vector_type(4)))  unsigned int  u32x4;

union Frag { bf16x16 bf; u32x4 q[2]; };

__device__ __forceinline__ unsigned short f2bf(float f) {
  unsigned int u = __float_as_uint(f);
  unsigned int r = u + 0x7fffu + ((u >> 16) & 1u);   // round-to-nearest-even
  return (unsigned short)(r >> 16);
}
__device__ __forceinline__ float siluf(float x) { return x / (1.f + __expf(-x)); }

// ---------------------------------------------------------------- f32 -> bf16
__global__ void __launch_bounds__(256) cvt_bf16_k(const float* __restrict__ in,
                                                  unsigned short* __restrict__ out,
                                                  int n) {
  int i = blockIdx.x * 256 + threadIdx.x;
  if (i < n) out[i] = f2bf(in[i]);
}

// ------------------------------------------------------- lengths from padding
__global__ void __launch_bounds__(256) lengths_k(const unsigned char* __restrict__ mask,
                                                 int* __restrict__ lengths) {
  const int b = blockIdx.x;
  int cnt = 0;
  for (int i = threadIdx.x; i < SEQ; i += 256) cnt += (mask[b * SEQ + i] != 0);
  __shared__ int r[256];
  r[threadIdx.x] = cnt; __syncthreads();
  for (int o = 128; o > 0; o >>= 1) {
    if (threadIdx.x < o) r[threadIdx.x] += r[threadIdx.x + o];
    __syncthreads();
  }
  if (threadIdx.x == 0) lengths[b] = SEQ - r[0];
}

// --------------------------------------------------------- input layernorm
__global__ void __launch_bounds__(256) layernorm_in_k(const float* __restrict__ x,
                                                      float* __restrict__ h,
                                                      unsigned short* __restrict__ hbf) {
  const int row = blockIdx.x;
  const float* xr = x + (size_t)row * D_MODEL;
  float v[4]; float s = 0.f, s2 = 0.f;
#pragma unroll
  for (int i = 0; i < 4; ++i) {
    v[i] = xr[threadIdx.x + i * 256];
    s += v[i]; s2 += v[i] * v[i];
  }
  __shared__ float rs[256], rq[256];
  rs[threadIdx.x] = s; rq[threadIdx.x] = s2; __syncthreads();
  for (int o = 128; o > 0; o >>= 1) {
    if (threadIdx.x < o) { rs[threadIdx.x] += rs[threadIdx.x + o]; rq[threadIdx.x] += rq[threadIdx.x + o]; }
    __syncthreads();
  }
  const float mean = rs[0] * (1.f / D_MODEL);
  const float var  = rq[0] * (1.f / D_MODEL) - mean * mean;
  const float inv  = rsqrtf(var + EPSV);
#pragma unroll
  for (int i = 0; i < 4; ++i) {
    const int c = threadIdx.x + i * 256;
    const float o = (v[i] - mean) * inv;
    h  [(size_t)row * D_MODEL + c] = o;
    hbf[(size_t)row * D_MODEL + c] = f2bf(o);
  }
}

// ---------------------------------------------------- flip h along L -> bf16
__global__ void __launch_bounds__(256) flip_bf16_k(const float* __restrict__ h,
                                                   const int* __restrict__ lengths,
                                                   unsigned short* __restrict__ hbf) {
  const size_t idx = (size_t)blockIdx.x * 256 + threadIdx.x;
  if (idx >= (size_t)ROWS * D_MODEL) return;
  const int c  = (int)(idx & (D_MODEL - 1));
  const int bl = (int)(idx >> 10);
  const int l  = bl & (SEQ - 1);
  const int b  = bl >> 10;
  const int len = lengths[b];
  const int pos = (l < len) ? (len - 1 - l) : l;
  hbf[idx] = f2bf(h[((size_t)(b * SEQ + pos)) * D_MODEL + c]);
}

// -------------------------------------------------------------- WMMA NT GEMM
// C[M,N] = A[M,K] * W[N,K]^T  (A,W bf16 K-major, C fp32).
// Each wave computes a 32x32 output macro-tile (2x2 WMMA tiles): every A/B
// fragment is reused twice -> 2 b128 loads per v_wmma_f32_16x16x32_bf16.
__global__ void __launch_bounds__(256) wmma_gemm_nt_k(const unsigned short* __restrict__ A,
                                                      const unsigned short* __restrict__ W,
                                                      float* __restrict__ C,
                                                      int M, int N, int K) {
  const int wave   = blockIdx.x * 8 + (threadIdx.x >> 5);
  const int lane   = threadIdx.x & 31;
  const int mpairs = M >> 5;
  const int npairs = N >> 5;
  if (wave >= mpairs * npairs) return;          // wave-uniform: EXEC stays all-1s for WMMA
  const int mp   = wave % mpairs;
  const int np   = wave / mpairs;
  const int half = lane >> 4;
  const int l16  = lane & 15;

  const unsigned short* a0r = A + (size_t)((mp << 5) + l16) * K;
  const unsigned short* a1r = a0r + (size_t)16 * K;
  const unsigned short* b0r = W + (size_t)((np << 5) + l16) * K;
  const unsigned short* b1r = b0r + (size_t)16 * K;

  f32x8 acc00 = {0.f,0.f,0.f,0.f,0.f,0.f,0.f,0.f};
  f32x8 acc01 = acc00, acc10 = acc00, acc11 = acc00;

  for (int k = 0; k < K; k += 32) {
    Frag a0, a1, b0, b1;
    // A 16x32 frag: lane(half,m): VGPR0-3 = K[half*8..+7], VGPR4-7 = K[16+half*8..+7]
    a0.q[0] = *(const u32x4*)(a0r + k + half * 8);
    a0.q[1] = *(const u32x4*)(a0r + k + 16 + half * 8);
    a1.q[0] = *(const u32x4*)(a1r + k + half * 8);
    a1.q[1] = *(const u32x4*)(a1r + k + 16 + half * 8);
    // B 32x16 frag: lane(half,n): VGPR0-3 = K[half*8..+7], VGPR4-7 = K[16+half*8..+7]
    b0.q[0] = *(const u32x4*)(b0r + k + half * 8);
    b0.q[1] = *(const u32x4*)(b0r + k + 16 + half * 8);
    b1.q[0] = *(const u32x4*)(b1r + k + half * 8);
    b1.q[1] = *(const u32x4*)(b1r + k + 16 + half * 8);
    acc00 = __builtin_amdgcn_wmma_f32_16x16x32_bf16(false, a0.bf, false, b0.bf, (short)0, acc00, false, false);
    acc01 = __builtin_amdgcn_wmma_f32_16x16x32_bf16(false, a0.bf, false, b1.bf, (short)0, acc01, false, false);
    acc10 = __builtin_amdgcn_wmma_f32_16x16x32_bf16(false, a1.bf, false, b0.bf, (short)0, acc10, false, false);
    acc11 = __builtin_amdgcn_wmma_f32_16x16x32_bf16(false, a1.bf, false, b1.bf, (short)0, acc11, false, false);
  }
  const int ncol = (np << 5) + l16;
#pragma unroll
  for (int r = 0; r < 8; ++r) {
    const size_t m0 = (size_t)((mp << 5) + r + (half << 3));
    const size_t m1 = m0 + 16;
    C[m0 * N + ncol]      = acc00[r];
    C[m0 * N + ncol + 16] = acc01[r];
    C[m1 * N + ncol]      = acc10[r];
    C[m1 * N + ncol + 16] = acc11[r];
  }
}

// ---------------------------------------------- depthwise causal conv + SiLU
__global__ void __launch_bounds__(256) conv_silu_k(const float* __restrict__ zx,
                                                   const float* __restrict__ cw,
                                                   const float* __restrict__ cb,
                                                   float* __restrict__ xBC) {
  const size_t idx = (size_t)blockIdx.x * 256 + threadIdx.x;
  if (idx >= (size_t)ROWS * CONV_DIM) return;
  const int c  = (int)(idx % CONV_DIM);
  const int bl = (int)(idx / CONV_DIM);
  const int l  = bl & (SEQ - 1);
  const int b  = bl >> 10;
  float acc = cb[c];
#pragma unroll
  for (int k = 0; k < D_CONV; ++k) {
    const int lk = l - (D_CONV - 1) + k;
    if (lk >= 0)
      acc += cw[c * D_CONV + k] * zx[((size_t)(b * SEQ + lk)) * D_IN_PROJ + D_INNER + c];
  }
  xBC[idx] = siluf(acc);
}

// -------------------------------------------- dt softplus and dA = exp(dt*A)
__global__ void __launch_bounds__(256) dt_k(const float* __restrict__ zx,
                                            const float* __restrict__ dt_bias,
                                            const float* __restrict__ A_log,
                                            float* __restrict__ dtp,
                                            float* __restrict__ dA) {
  const int idx = blockIdx.x * 256 + threadIdx.x;
  if (idx >= ROWS * NHEADS) return;
  const int hh  = idx & (NHEADS - 1);
  const int row = idx >> 5;
  const float v  = zx[(size_t)row * D_IN_PROJ + D_INNER + CONV_DIM + hh] + dt_bias[hh];
  const float sp = (v > 20.f) ? v : log1pf(__expf(v));
  dtp[idx] = sp;
  dA[idx]  = __expf(-__expf(A_log[hh]) * sp);
}

// ------------------------------------------------- selective scan, one (b,h)
// 256 threads: p = tid>>2 (0..63), g = tid&3 covers n in [g*32,(g+1)*32).
// Register-staged, double-buffered LDS pipeline: loads for step t+1 are issued
// before the (single) barrier of step t, hiding global latency behind compute.
__global__ void __launch_bounds__(256) scan_k(const float* __restrict__ xBC,
                                              const float* __restrict__ dtp,
                                              const float* __restrict__ dA,
                                              const float* __restrict__ Dpar,
                                              float* __restrict__ y) {
  const int b    = blockIdx.x >> 5;
  const int head = blockIdx.x & (NHEADS - 1);
  const int tid  = threadIdx.x;
  const int p    = tid >> 2;
  const int g    = tid & 3;
  __shared__ float Bs[2][D_STATE], Cs[2][D_STATE], xs[2][HEADDIM];
  float s[32];
#pragma unroll
  for (int j = 0; j < 32; ++j) s[j] = 0.f;
  const float Dh = Dpar[head];
  const float* base = xBC + (size_t)b * SEQ * CONV_DIM;
  const float* dAb  = dA  + (size_t)b * SEQ * NHEADS + head;
  const float* dtb  = dtp + (size_t)b * SEQ * NHEADS + head;

  // ---- prologue: stage t = 0 into registers
  float rBC, rx = 0.f, rdA, rdt;
  {
    const float* row = base;
    rBC = (tid < 128) ? row[D_INNER + tid] : row[D_INNER + D_STATE + (tid - 128)];
    if (tid < HEADDIM) rx = row[head * HEADDIM + tid];
    rdA = dAb[0]; rdt = dtb[0];
  }

  int buf = 0;
  for (int t = 0; t < SEQ; ++t) {
    // commit staged registers for step t
    if (tid < 128) Bs[buf][tid] = rBC;
    else           Cs[buf][tid - 128] = rBC;
    if (tid < HEADDIM) xs[buf][tid] = rx;
    const float dA_t = rdA;
    const float dt_t = rdt;

    // stage step t+1 (loads overlap with compute below)
    if (t + 1 < SEQ) {
      const float* row = base + (size_t)(t + 1) * CONV_DIM;
      rBC = (tid < 128) ? row[D_INNER + tid] : row[D_INNER + D_STATE + (tid - 128)];
      if (tid < HEADDIM) rx = row[head * HEADDIM + tid];
      rdA = dAb[(size_t)(t + 1) * NHEADS];
      rdt = dtb[(size_t)(t + 1) * NHEADS];
    }
    __syncthreads();

    const float xb = dt_t * xs[buf][p];
    const float* Bp = Bs[buf] + g * 32;
    const float* Cp = Cs[buf] + g * 32;
    float acc = 0.f;
#pragma unroll
    for (int j = 0; j < 32; ++j) {
      s[j] = fmaf(s[j], dA_t, xb * Bp[j]);
      acc  = fmaf(s[j], Cp[j], acc);
    }
    acc += __shfl_xor(acc, 1, 32);
    acc += __shfl_xor(acc, 2, 32);
    if (g == 0)
      y[((size_t)(b * SEQ + t)) * D_INNER + head * HEADDIM + p] = acc + Dh * xs[buf][p];
    buf ^= 1;
  }
}

// ---------------------------------------------- gated RMSNorm -> bf16 output
__global__ void __launch_bounds__(256) grms_k(const float* __restrict__ y,
                                              const float* __restrict__ zx,
                                              const float* __restrict__ nw,
                                              unsigned short* __restrict__ Ybf) {
  const int row = blockIdx.x;
  float g[8]; float s2 = 0.f;
#pragma unroll
  for (int i = 0; i < 8; ++i) {
    const int c = threadIdx.x + i * 256;
    const float z  = zx[(size_t)row * D_IN_PROJ + c];
    const float gg = y[(size_t)row * D_INNER + c] * siluf(z);
    g[i] = gg; s2 += gg * gg;
  }
  __shared__ float r[256];
  r[threadIdx.x] = s2; __syncthreads();
  for (int o = 128; o > 0; o >>= 1) {
    if (threadIdx.x < o) r[threadIdx.x] += r[threadIdx.x + o];
    __syncthreads();
  }
  const float inv = rsqrtf(r[0] * (1.f / D_INNER) + EPSV);
#pragma unroll
  for (int i = 0; i < 8; ++i) {
    const int c = threadIdx.x + i * 256;
    Ybf[(size_t)row * D_INNER + c] = f2bf(g[i] * inv * nw[c]);
  }
}

// ---------------------- combine fwd + flipped-back rev, average, layernorm
__global__ void __launch_bounds__(256) combine_ln_k(const float* __restrict__ outf,
                                                    const float* __restrict__ outr,
                                                    const int* __restrict__ lengths,
                                                    float* __restrict__ out) {
  const int row = blockIdx.x;
  const int b = row >> 10;
  const int l = row & (SEQ - 1);
  const int len = lengths[b];
  const int pos = (l < len) ? (len - 1 - l) : l;
  const float* pf = outf + (size_t)row * D_MODEL;
  const float* pr = outr + ((size_t)(b * SEQ + pos)) * D_MODEL;
  float v[4]; float s = 0.f, s2 = 0.f;
#pragma unroll
  for (int i = 0; i < 4; ++i) {
    const int c = threadIdx.x + i * 256;
    v[i] = 0.5f * (pf[c] + pr[c]);
    s += v[i]; s2 += v[i] * v[i];
  }
  __shared__ float rs[256], rq[256];
  rs[threadIdx.x] = s; rq[threadIdx.x] = s2; __syncthreads();
  for (int o = 128; o > 0; o >>= 1) {
    if (threadIdx.x < o) { rs[threadIdx.x] += rs[threadIdx.x + o]; rq[threadIdx.x] += rq[threadIdx.x + o]; }
    __syncthreads();
  }
  const float mean = rs[0] * (1.f / D_MODEL);
  const float var  = rq[0] * (1.f / D_MODEL) - mean * mean;
  const float inv  = rsqrtf(var + EPSV);
#pragma unroll
  for (int i = 0; i < 4; ++i) {
    const int c = threadIdx.x + i * 256;
    out[(size_t)row * D_MODEL + c] = (v[i] - mean) * inv;
  }
}

// ============================================================================
extern "C" void kernel_launch(void* const* d_in, const int* in_sizes, int n_in,
                              void* d_out, int out_size, void* d_ws, size_t ws_size,
                              hipStream_t stream) {
  (void)in_sizes; (void)n_in; (void)out_size; (void)ws_size;

  const float*         x_in  = (const float*)d_in[0];
  const unsigned char* mask  = (const unsigned char*)d_in[1];
  const float*         in_w  = (const float*)d_in[2];
  const float*         out_w = (const float*)d_in[3];
  const float* conv_w[2] = {(const float*)d_in[4],  (const float*)d_in[10]};
  const float* conv_b[2] = {(const float*)d_in[5],  (const float*)d_in[11]};
  const float* dt_b[2]   = {(const float*)d_in[6],  (const float*)d_in[12]};
  const float* A_log[2]  = {(const float*)d_in[7],  (const float*)d_in[13]};
  const float* Dp[2]     = {(const float*)d_in[8],  (const float*)d_in[14]};
  const float* nw[2]     = {(const float*)d_in[9],  (const float*)d_in[15]};

  char* cur = (char*)d_ws;
  auto take = [&](size_t bytes) -> void* {
    void* p = (void*)cur;
    cur += (bytes + 255) & ~(size_t)255;
    return p;
  };
  float*          h      = (float*)         take((size_t)ROWS * D_MODEL   * 4);
  unsigned short* hbf    = (unsigned short*)take((size_t)ROWS * D_MODEL   * 2);
  unsigned short* Winbf  = (unsigned short*)take((size_t)D_IN_PROJ * D_MODEL * 2);
  unsigned short* Woutbf = (unsigned short*)take((size_t)D_MODEL * D_INNER * 2);
  float*          zx     = (float*)         take((size_t)ROWS * D_IN_PROJ * 4);
  float*          xBC    = (float*)         take((size_t)ROWS * CONV_DIM  * 4);
  float*          dtp    = (float*)         take((size_t)ROWS * NHEADS    * 4);
  float*          dAb    = (float*)         take((size_t)ROWS * NHEADS    * 4);
  float*          yb     = (float*)         take((size_t)ROWS * D_INNER   * 4);
  unsigned short* Ybf    = (unsigned short*)take((size_t)ROWS * D_INNER   * 2);
  float*          outd0  = (float*)         take((size_t)ROWS * D_MODEL   * 4);
  float*          outd1  = (float*)         take((size_t)ROWS * D_MODEL   * 4);
  int*            lens   = (int*)           take(256);
  float* outd[2] = {outd0, outd1};

  const int nInW  = D_IN_PROJ * D_MODEL;
  const int nOutW = D_MODEL * D_INNER;
  cvt_bf16_k<<<(nInW  + 255) / 256, 256, 0, stream>>>(in_w,  Winbf,  nInW);
  cvt_bf16_k<<<(nOutW + 255) / 256, 256, 0, stream>>>(out_w, Woutbf, nOutW);
  lengths_k<<<BSZ, 256, 0, stream>>>(mask, lens);
  layernorm_in_k<<<ROWS, 256, 0, stream>>>(x_in, h, hbf);

  for (int dir = 0; dir < 2; ++dir) {
    if (dir == 1)
      flip_bf16_k<<<(ROWS * D_MODEL) / 256, 256, 0, stream>>>(h, lens, hbf);

    // in_proj: [4096 x 4384] = hbf[4096 x 1024] * Winbf[4384 x 1024]^T
    wmma_gemm_nt_k<<<((ROWS / 32) * (D_IN_PROJ / 32)) / 8, 256, 0, stream>>>(
        hbf, Winbf, zx, ROWS, D_IN_PROJ, D_MODEL);

    conv_silu_k<<<((size_t)ROWS * CONV_DIM + 255) / 256, 256, 0, stream>>>(
        zx, conv_w[dir], conv_b[dir], xBC);
    dt_k<<<(ROWS * NHEADS) / 256, 256, 0, stream>>>(zx, dt_b[dir], A_log[dir], dtp, dAb);
    scan_k<<<BSZ * NHEADS, 256, 0, stream>>>(xBC, dtp, dAb, Dp[dir], yb);
    grms_k<<<ROWS, 256, 0, stream>>>(yb, zx, nw[dir], Ybf);

    // out_proj: [4096 x 1024] = Ybf[4096 x 2048] * Woutbf[1024 x 2048]^T
    wmma_gemm_nt_k<<<((ROWS / 32) * (D_MODEL / 32)) / 8, 256, 0, stream>>>(
        Ybf, Woutbf, outd[dir], ROWS, D_MODEL, D_INNER);
  }

  combine_ln_k<<<ROWS, 256, 0, stream>>>(outd[0], outd[1], lens, (float*)d_out);
}
